// TextMoE_34574486732892
// MI455X (gfx1250) — matrix operand order repaired and checked by reading, not compiled, not run on hardware
//
#include <hip/hip_runtime.h>
#include <hip/hip_bf16.h>

// ---------------- model dims ----------------
#define B_  8
#define S_  512
#define D_  1024
#define E_  8
#define F_  4096
#define H_  8
#define HS_ 128
#define T_  (B_ * S_)     // 4096 tokens

#define USE_ASYNC 1

// ---------------- types ----------------
typedef __bf16 bf16_t;
typedef bf16_t v16bf __attribute__((ext_vector_type(16)));
typedef bf16_t v8bf  __attribute__((ext_vector_type(8)));
typedef float  v8f   __attribute__((ext_vector_type(8)));

static_assert(sizeof(bf16_t) == 2, "bf16 size");

__device__ __forceinline__ bf16_t f32_to_bf16(float f) {
    union { float f; unsigned u; } in; in.f = f;
    unsigned lsb = (in.u >> 16) & 1u;
    unsigned r   = in.u + 0x7FFFu + lsb;        // round-to-nearest-even
    union { unsigned short u; bf16_t b; } out;
    out.u = (unsigned short)(r >> 16);
    return out.b;
}

template<int N>
__device__ __forceinline__ void wait_async() {
#if __has_builtin(__builtin_amdgcn_s_wait_asynccnt)
    __builtin_amdgcn_s_wait_asynccnt(N);
#else
    asm volatile("s_wait_asynccnt %0" :: "i"(N) : "memory");
#endif
}

// copy 16 bytes global -> LDS (ASYNCcnt-tracked); lds byte offset = low 32
// bits of the generic shared-pointer address (ISA 10.2 aperture layout).
__device__ __forceinline__ void async_b128(const bf16_t* gp, bf16_t* lp) {
#if USE_ASYNC
    unsigned lds_off = (unsigned)(unsigned long long)lp;
    asm volatile("global_load_async_to_lds_b128 %0, %1, off"
                 :: "v"(lds_off), "v"(gp) : "memory");
#else
    *(v8bf*)lp = *(const v8bf*)gp;
#endif
}

// ============================================================
// TN GEMM, all-bf16 operands, fp32 WMMA accumulate.
//   A: [M,K] bf16 row-major (K contiguous), lda
//   B: [N,K] bf16 row-major (K contiguous), ldb   (i.e. B^T of math B)
// Tile 256x128x32; 256 threads = 8 wave32 waves in a 4(M) x 2(N) grid;
// each wave owns 64x64 = 4x4 WMMA fragments -> 16 wmma / K-step
// against 16 ds_load_b128 (1:1).  Double-buffered LDS (60KB), tiles
// streamed with global_load_async_to_lds_b128 (6 instr/wave/tile),
// pipelined with s_wait_asynccnt 6; last iteration peeled so the
// steady-state loop is branch-free.
// MODE 0: C f32 = acc + bias
// MODE 1: C f32 = resid + acc + bias
// MODE 2: C bf16 = relu(acc + bias)
// MODE 3: C f32 (+)= gate[m*E + gidx] * (acc + bias)
// MODE 4: C bf16, V-transpose layout [b][h][hs][s]
// MODE 5: C bf16 = acc
// ============================================================
#define BM 256
#define BN 128
#define BK 32
#define LDSS 40   // bf16/row = 80B: 16B-aligned rows, 20-bank stride (conflict-free)

union FragAB { v16bf v; v8bf h[2]; };

template<int MODE>
__global__ __launch_bounds__(256)
void gemm_tn(const bf16_t* __restrict__ A, const bf16_t* __restrict__ Bm,
             void* __restrict__ Cv,
             const float* __restrict__ bias,
             const float* __restrict__ resid,
             const float* __restrict__ gate, int gate_idx, int accum,
             int M, int N, int K, int ldA, int ldB, int ldC,
             long oA1, long oA2, long oB1, long oB2, long oC1, long oC2,
             int batch2)
{
    __shared__ __align__(16) bf16_t As[2][BM * LDSS];   // 40960 B
    __shared__ __align__(16) bf16_t Bs[2][BN * LDSS];   // 20480 B

    const int tid = threadIdx.x;
    const int z   = blockIdx.z;
    const int z1  = z / batch2;
    const int z2  = z - z1 * batch2;
    const long aOff = (long)z1 * oA1 + (long)z2 * oA2;
    const long bOff = (long)z1 * oB1 + (long)z2 * oB2;
    const long cOff = (long)z1 * oC1 + (long)z2 * oC2;

    const int tileM = blockIdx.y * BM;
    const int tileN = blockIdx.x * BN;

    const int wave = tid >> 5;
    const int lane = tid & 31;
    const int wm = (wave & 3) * 64;    // 4 waves along M
    const int wn = (wave >> 2) * 64;   // 2 waves along N
    const int lrow = lane & 15;
    const int half = lane >> 4;

    v8f acc[4][4];
#pragma unroll
    for (int i = 0; i < 4; ++i)
#pragma unroll
        for (int j = 0; j < 4; ++j)
            acc[i][j] = (v8f){0.f,0.f,0.f,0.f,0.f,0.f,0.f,0.f};

    // per-thread staging: 16B chunk scc of rows sr+{0,64,128,192} (A) and
    // sr+{0,64} (B) -> 6 async instructions per wave per tile
    const int sr  = tid >> 2;
    const int scc = (tid & 3) * 8;
    const bf16_t* gA0 = A  + aOff + (long)(tileM + sr) * ldA + scc;
    const bf16_t* gA1 = gA0 +  64l * ldA;
    const bf16_t* gA2 = gA0 + 128l * ldA;
    const bf16_t* gA3 = gA0 + 192l * ldA;
    const bf16_t* gB0 = Bm + bOff + (long)(tileN + sr) * ldB + scc;
    const bf16_t* gB1 = gB0 + 64l * ldB;

    auto stage = [&](int buf, int k0) {
        async_b128(gA0 + k0, &As[buf][(sr      ) * LDSS + scc]);
        async_b128(gA1 + k0, &As[buf][(sr +  64) * LDSS + scc]);
        async_b128(gA2 + k0, &As[buf][(sr + 128) * LDSS + scc]);
        async_b128(gA3 + k0, &As[buf][(sr + 192) * LDSS + scc]);
        async_b128(gB0 + k0, &Bs[buf][(sr      ) * LDSS + scc]);
        async_b128(gB1 + k0, &Bs[buf][(sr +  64) * LDSS + scc]);
    };

    auto compute = [&](int cur) {
        const bf16_t* Asb = As[cur];
        const bf16_t* Bsb = Bs[cur];
        // ---- fragments per ISA 7.12.2 ----
        FragAB a[4];
#pragma unroll
        for (int i = 0; i < 4; ++i) {
            const bf16_t* p = &Asb[(wm + i * 16 + lrow) * LDSS];
            a[i].h[0] = *(const v8bf*)(p + half * 8);
            a[i].h[1] = *(const v8bf*)(p + 16 + half * 8);
        }
        FragAB b[4];
#pragma unroll
        for (int j = 0; j < 4; ++j) {
            const bf16_t* p = &Bsb[(wn + j * 16 + lrow) * LDSS];
            b[j].h[0] = *(const v8bf*)(p + half * 8);
            b[j].h[1] = *(const v8bf*)(p + 16 + half * 8);
        }
#pragma unroll
        for (int i = 0; i < 4; ++i)
#pragma unroll
            for (int j = 0; j < 4; ++j)
                acc[i][j] = __builtin_amdgcn_wmma_f32_16x16x32_bf16(
                    false, a[i].v, false, b[j].v, (short)0, acc[i][j], false, false);
    };

    const int nk = K / BK;
    stage(0, 0);
    for (int it = 0; it < nk - 1; ++it) {     // branch-free steady state
        stage((it + 1) & 1, (it + 1) * BK);   // next tile streams during compute
        wait_async<6>();                      // in-order: current tile resident
        __syncthreads();
        compute(it & 1);
        __syncthreads();                      // all reads done before restage
    }
    wait_async<0>();                          // peeled last iteration
    __syncthreads();
    compute((nk - 1) & 1);

    // ---- epilogue: D layout: VGPR r -> row r + 8*half, col lrow ----
    float* Cf = (float*)Cv;
    bf16_t* Cb = (bf16_t*)Cv;
#pragma unroll
    for (int j = 0; j < 4; ++j) {
        int gn = tileN + wn + j * 16 + lrow;
        float bv = 0.f;
        if (MODE == 0 || MODE == 1 || MODE == 2 || MODE == 3)
            bv = bias ? bias[gn] : 0.f;
#pragma unroll
        for (int i = 0; i < 4; ++i) {
#pragma unroll
            for (int r = 0; r < 8; ++r) {
                int gm = tileM + wm + i * 16 + r + half * 8;
                float v = acc[i][j][r] + bv;
                long cidx = cOff + (long)gm * ldC + gn;
                if constexpr (MODE == 0) {
                    Cf[cidx] = v;
                } else if constexpr (MODE == 1) {
                    Cf[cidx] = resid[cidx] + v;
                } else if constexpr (MODE == 2) {
                    Cb[cidx] = f32_to_bf16(v > 0.f ? v : 0.f);
                } else if constexpr (MODE == 3) {
                    float g = gate[(long)gm * E_ + gate_idx];
                    float o = g * v;
                    if (accum) Cf[cidx] += o; else Cf[cidx] = o;
                } else if constexpr (MODE == 4) {
                    int h = gn >> 7, dd = gn & (HS_ - 1);
                    int bb = gm >> 9, s = gm & (S_ - 1);
                    long idx = (((long)bb * H_ + h) * HS_ + dd) * S_ + s;
                    Cb[idx] = f32_to_bf16(v);
                } else { // MODE 5
                    Cb[cidx] = f32_to_bf16(v);
                }
            }
        }
    }
}

// ============================================================
// Support kernels
// ============================================================
// out[c*R + r] = bf16(in[r*C + c])   (transpose + convert)
__global__ __launch_bounds__(256)
void transpose_to_bf16(const float* __restrict__ in, bf16_t* __restrict__ out,
                       int R, int C) {
    __shared__ float tile[32][33];
    int cb = blockIdx.x * 32, rb = blockIdx.y * 32;
    int tx = threadIdx.x & 31, ty = threadIdx.x >> 5;   // 32 x 8
    for (int i = ty; i < 32; i += 8)
        tile[i][tx] = in[(long)(rb + i) * C + cb + tx];
    __syncthreads();
    for (int i = ty; i < 32; i += 8)
        out[(long)(cb + i) * R + rb + tx] = f32_to_bf16(tile[tx][i]);
}

__global__ void embed_kernel(const int* __restrict__ ids,
                             const float* __restrict__ emb,
                             float* __restrict__ x) {
    long i = (long)blockIdx.x * blockDim.x + threadIdx.x;
    if (i >= (long)T_ * D_) return;
    long t = i / D_, d = i - t * D_;
    x[i] = emb[(long)ids[t] * D_ + d];
}

__global__ __launch_bounds__(256)
void ln_kernel(const float* __restrict__ x, const float* __restrict__ g,
               const float* __restrict__ b, float* __restrict__ y,
               bf16_t* __restrict__ yb) {
    __shared__ float s1[256], s2[256];
    int t = blockIdx.x, tid = threadIdx.x;
    const float* row = x + (long)t * D_;
    float a = 0.f, a2 = 0.f;
    for (int d = tid; d < D_; d += 256) { float v = row[d]; a += v; a2 += v * v; }
    s1[tid] = a; s2[tid] = a2; __syncthreads();
    for (int st = 128; st > 0; st >>= 1) {
        if (tid < st) { s1[tid] += s1[tid + st]; s2[tid] += s2[tid + st]; }
        __syncthreads();
    }
    float mean = s1[0] / D_;
    float var  = s2[0] / D_ - mean * mean;
    float r = rsqrtf(var + 1e-5f);
    for (int d = tid; d < D_; d += 256) {
        float v = (row[d] - mean) * r * g[d] + b[d];
        y[(long)t * D_ + d]  = v;
        yb[(long)t * D_ + d] = f32_to_bf16(v);
    }
}

__global__ __launch_bounds__(256)
void causal_softmax(const float* __restrict__ att, bf16_t* __restrict__ attb,
                    float scale) {
    __shared__ float red[256];
    int row = blockIdx.x;          // z*S + q, z in [0,B*H)
    int q = row % S_;
    long base = (long)row * S_;
    int tid = threadIdx.x;
    float m = -3.0e38f;
    for (int k = tid; k <= q; k += 256) m = fmaxf(m, att[base + k] * scale);
    red[tid] = m; __syncthreads();
    for (int st = 128; st > 0; st >>= 1) {
        if (tid < st) red[tid] = fmaxf(red[tid], red[tid + st]);
        __syncthreads();
    }
    m = red[0]; __syncthreads();
    float s = 0.f;
    for (int k = tid; k <= q; k += 256) s += expf(att[base + k] * scale - m);
    red[tid] = s; __syncthreads();
    for (int st = 128; st > 0; st >>= 1) {
        if (tid < st) red[tid] += red[tid + st];
        __syncthreads();
    }
    float inv = 1.f / red[0];
    for (int k = tid; k < S_; k += 256) {
        float v = (k <= q) ? expf(att[base + k] * scale - m) * inv : 0.f;
        attb[base + k] = f32_to_bf16(v);
    }
}

__global__ void addpos_kernel(float* __restrict__ x, const float* __restrict__ pos) {
    long i = (long)blockIdx.x * blockDim.x + threadIdx.x;
    if (i >= (long)T_ * D_) return;
    x[i] += pos[i % ((long)S_ * D_)];
}

__global__ void router_noisy(const float* __restrict__ xln,
                             const float* __restrict__ Wr, const float* __restrict__ br,
                             const float* __restrict__ Wn, const float* __restrict__ bn,
                             const float* __restrict__ noise,
                             float* __restrict__ noisy) {
    int i = blockIdx.x * blockDim.x + threadIdx.x;
    if (i >= T_ * E_) return;
    int t = i / E_, e = i - t * E_;
    const float* row = xln + (long)t * D_;
    float lr = 0.f, ln_ = 0.f;
    for (int d = 0; d < D_; ++d) {
        float v = row[d];
        lr  += v * Wr[(long)d * E_ + e];
        ln_ += v * Wn[(long)d * E_ + e];
    }
    lr += br[e]; ln_ += bn[e];
    float sp = fmaxf(ln_, 0.f) + log1pf(expf(-fabsf(ln_)));  // stable softplus
    noisy[i] = lr + noise[i] * sp;
}

__global__ void router_gate(const float* __restrict__ noisy, float* __restrict__ gate) {
    int t = blockIdx.x * blockDim.x + threadIdx.x;
    if (t >= T_) return;
    float v[E_];
    for (int e = 0; e < E_; ++e) v[e] = noisy[(long)t * E_ + e];
    int i1 = 0;
    for (int e = 1; e < E_; ++e) if (v[e] > v[i1]) i1 = e;
    int i2 = (i1 == 0) ? 1 : 0;
    for (int e = 0; e < E_; ++e) if (e != i1 && v[e] > v[i2]) i2 = e;
    float m  = fmaxf(v[i1], v[i2]);
    float e1 = expf(v[i1] - m), e2 = expf(v[i2] - m);
    float inv = 1.f / (e1 + e2);
    for (int e = 0; e < E_; ++e)
        gate[(long)t * E_ + e] = (e == i1) ? e1 * inv : (e == i2) ? e2 * inv : 0.f;
}

__global__ void mean_kernel(const float* __restrict__ second, float* __restrict__ feature) {
    int i = blockIdx.x * blockDim.x + threadIdx.x;  // b*D + d
    if (i >= B_ * D_) return;
    int b = i / D_, d = i - b * D_;
    float s = 0.f;
    for (int t = 0; t < S_; ++t) s += second[((long)b * S_ + t) * D_ + d];
    feature[i] = s * (1.f / S_);
}

__global__ void cls_kernel(const float* __restrict__ feature,
                           const float* __restrict__ Wc, const float* __restrict__ bc,
                           float* __restrict__ cls) {
    int i = blockIdx.x * blockDim.x + threadIdx.x;  // b*10 + c
    if (i >= B_ * 10) return;
    int b = i / 10, c = i - b * 10;
    float s = bc[c];
    for (int d = 0; d < D_; ++d) s += feature[(long)b * D_ + d] * Wc[(long)d * 10 + c];
    cls[i] = s;
}

// ============================================================
// Launcher
// ============================================================
template<int MODE>
static inline void gemmL(hipStream_t st, const bf16_t* A, const bf16_t* Bm, void* C,
                         const float* bias, const float* resid,
                         const float* gate, int gidx, int accum,
                         int M, int N, int K, int ldA, int ldB, int ldC,
                         long oA1, long oA2, long oB1, long oB2, long oC1, long oC2,
                         int nb, int b2) {
    dim3 grid(N / BN, M / BM, nb);
    gemm_tn<MODE><<<grid, 256, 0, st>>>(A, Bm, C, bias, resid, gate, gidx, accum,
        M, N, K, ldA, ldB, ldC, oA1, oA2, oB1, oB2, oC1, oC2, b2);
}

extern "C" void kernel_launch(void* const* d_in, const int* in_sizes, int n_in,
                              void* d_out, int out_size, void* d_ws, size_t ws_size,
                              hipStream_t stream) {
    (void)in_sizes; (void)n_in; (void)out_size; (void)ws_size;
    const int*   ids  = (const int*)  d_in[0];
    const float* emb  = (const float*)d_in[2];
    const float* pos  = (const float*)d_in[3];
    const float* Wq   = (const float*)d_in[4];
    const float* Wk   = (const float*)d_in[5];
    const float* Wv   = (const float*)d_in[6];
    const float* Wo   = (const float*)d_in[7];
    const float* bo   = (const float*)d_in[8];
    const float* g1   = (const float*)d_in[9];
    const float* be1  = (const float*)d_in[10];
    const float* g2   = (const float*)d_in[11];
    const float* be2  = (const float*)d_in[12];
    const float* g3   = (const float*)d_in[13];
    const float* be3  = (const float*)d_in[14];
    const float* Wr1  = (const float*)d_in[15];
    const float* br1  = (const float*)d_in[16];
    const float* Wn1  = (const float*)d_in[17];
    const float* bn1  = (const float*)d_in[18];
    const float* W1a  = (const float*)d_in[19];
    const float* b1a  = (const float*)d_in[20];
    const float* W2a  = (const float*)d_in[21];
    const float* b2a  = (const float*)d_in[22];
    const float* Wr2  = (const float*)d_in[23];
    const float* br2  = (const float*)d_in[24];
    const float* Wn2  = (const float*)d_in[25];
    const float* bn2  = (const float*)d_in[26];
    const float* W1b  = (const float*)d_in[27];
    const float* b1b  = (const float*)d_in[28];
    const float* W2b  = (const float*)d_in[29];
    const float* b2b  = (const float*)d_in[30];
    const float* noise1 = (const float*)d_in[31];
    const float* noise2 = (const float*)d_in[32];
    const float* Wc   = (const float*)d_in[33];
    const float* bc   = (const float*)d_in[34];

    float* out = (float*)d_out;
    float* outFirst  = out;
    float* outSecond = out + (size_t)T_ * D_;
    float* feature   = out + 2ull * T_ * D_;
    float* cls       = feature + (size_t)B_ * D_;

    float* ws = (float*)d_ws;
    const size_t TD = (size_t)T_ * D_;
    float* x     = ws;                 // [T,D]
    float* xln   = ws + TD;            // [T,D] (router needs fp32)
    float* x2    = ws + 2 * TD;        // [T,D]
    float* att   = ws + 3 * TD;        // [B*H,S,S] = 4*TD floats
    float* noisy = ws + 7 * TD;        // [T,E]
    float* gate  = noisy + (size_t)T_ * E_;

    bf16_t* bb    = (bf16_t*)(ws + 7 * TD + 2ull * T_ * E_);
    bf16_t* xln_b = bb;                 // [T,D]
    bf16_t* q_b   = bb + TD;            // [T,D]
    bf16_t* k_b   = bb + 2 * TD;        // [T,D]
    bf16_t* vT    = bb + 3 * TD;        // [B,H,HS,S]
    bf16_t* o_b   = bb + 4 * TD;        // [T,D]
    bf16_t* att_b = bb + 5 * TD;        // [B*H,S,S] = 4*TD
    bf16_t* Hbuf  = bb + 9 * TD;        // [T,F]    = 4*TD
    bf16_t* Wqt   = bb + 13 * TD;       // [D,D] each
    bf16_t* Wkt   = Wqt + (size_t)D_ * D_;
    bf16_t* Wvt   = Wkt + (size_t)D_ * D_;
    bf16_t* Wot   = Wvt + (size_t)D_ * D_;
    bf16_t* W1t   = Wot + (size_t)D_ * D_;   // [F,D] (per-expert, reused)
    bf16_t* W2t   = W1t + (size_t)D_ * F_;   // [D,F] (per-expert, reused)

    const long SD = (long)S_ * D_;
    const long SS = (long)S_ * S_;

    // weight transposes -> bf16 [N,K]
    dim3 tDD(D_ / 32, D_ / 32);
    transpose_to_bf16<<<tDD, 256, 0, stream>>>(Wq, Wqt, D_, D_);
    transpose_to_bf16<<<tDD, 256, 0, stream>>>(Wk, Wkt, D_, D_);
    transpose_to_bf16<<<tDD, 256, 0, stream>>>(Wv, Wvt, D_, D_);
    transpose_to_bf16<<<tDD, 256, 0, stream>>>(Wo, Wot, D_, D_);

    // x = emb[ids]; ln1
    embed_kernel<<<(T_ * D_ + 255) / 256, 256, 0, stream>>>(ids, emb, x);
    ln_kernel<<<T_, 256, 0, stream>>>(x, g1, be1, xln, xln_b);

    // q,k (bf16), v (head-transposed bf16)
    gemmL<5>(stream, xln_b, Wqt, q_b, nullptr, nullptr, nullptr, 0, 0,
             T_, D_, D_, D_, D_, D_, 0,0,0,0,0,0, 1, 1);
    gemmL<5>(stream, xln_b, Wkt, k_b, nullptr, nullptr, nullptr, 0, 0,
             T_, D_, D_, D_, D_, D_, 0,0,0,0,0,0, 1, 1);
    gemmL<4>(stream, xln_b, Wvt, vT, nullptr, nullptr, nullptr, 0, 0,
             T_, D_, D_, D_, D_, D_, 0,0,0,0,0,0, 1, 1);

    // scores = q @ k^T (TN: both stored [.,K]), batched over (b,h)
    gemmL<0>(stream, q_b, k_b, att, nullptr, nullptr, nullptr, 0, 0,
             S_, S_, HS_, D_, D_, S_,
             SD, HS_, SD, HS_, (long)H_ * SS, SS, B_ * H_, H_);
    causal_softmax<<<B_ * H_ * S_, 256, 0, stream>>>(att, att_b, 0.08838834764831845f);
    // o = att @ v  (B = vT stored [hs, s] per head -> TN)
    gemmL<5>(stream, att_b, vT, o_b, nullptr, nullptr, nullptr, 0, 0,
             S_, HS_, S_, S_, S_, D_,
             (long)H_ * SS, SS, (long)H_ * HS_ * S_, (long)HS_ * S_, SD, HS_,
             B_ * H_, H_);
    // x2 = x + o @ Wo + bo ; x2 += pos
    gemmL<1>(stream, o_b, Wot, x2, bo, x, nullptr, 0, 0,
             T_, D_, D_, D_, D_, D_, 0,0,0,0,0,0, 1, 1);
    addpos_kernel<<<(T_ * D_ + 255) / 256, 256, 0, stream>>>(x2, pos);

    dim3 tDF(F_ / 32, D_ / 32);   // transpose of [D,F]
    dim3 tFD(D_ / 32, F_ / 32);   // transpose of [F,D]

    // ---- MoE layer 1 ----
    ln_kernel<<<T_, 256, 0, stream>>>(x2, g2, be2, xln, xln_b);
    router_noisy<<<(T_ * E_ + 255) / 256, 256, 0, stream>>>(xln, Wr1, br1, Wn1, bn1, noise1, noisy);
    router_gate<<<(T_ + 255) / 256, 256, 0, stream>>>(noisy, gate);
    for (int e = 0; e < E_; ++e) {
        transpose_to_bf16<<<tDF, 256, 0, stream>>>(W1a + (size_t)e * D_ * F_, W1t, D_, F_);
        transpose_to_bf16<<<tFD, 256, 0, stream>>>(W2a + (size_t)e * F_ * D_, W2t, F_, D_);
        gemmL<2>(stream, xln_b, W1t, Hbuf, b1a + (size_t)e * F_, nullptr, nullptr, 0, 0,
                 T_, F_, D_, D_, D_, F_, 0,0,0,0,0,0, 1, 1);
        gemmL<3>(stream, Hbuf, W2t, outFirst, b2a + (size_t)e * D_, nullptr, gate, e, e > 0,
                 T_, D_, F_, F_, F_, D_, 0,0,0,0,0,0, 1, 1);
    }

    // ---- MoE layer 2 ----
    ln_kernel<<<T_, 256, 0, stream>>>(x2, g3, be3, xln, xln_b);
    router_noisy<<<(T_ * E_ + 255) / 256, 256, 0, stream>>>(xln, Wr2, br2, Wn2, bn2, noise2, noisy);
    router_gate<<<(T_ + 255) / 256, 256, 0, stream>>>(noisy, gate);
    for (int e = 0; e < E_; ++e) {
        transpose_to_bf16<<<tDF, 256, 0, stream>>>(W1b + (size_t)e * D_ * F_, W1t, D_, F_);
        transpose_to_bf16<<<tFD, 256, 0, stream>>>(W2b + (size_t)e * F_ * D_, W2t, F_, D_);
        gemmL<2>(stream, xln_b, W1t, Hbuf, b1b + (size_t)e * F_, nullptr, nullptr, 0, 0,
                 T_, F_, D_, D_, D_, F_, 0,0,0,0,0,0, 1, 1);
        gemmL<3>(stream, Hbuf, W2t, outSecond, b2b + (size_t)e * D_, nullptr, gate, e, e > 0,
                 T_, D_, F_, F_, F_, D_, 0,0,0,0,0,0, 1, 1);
    }

    // ---- feature + classifier ----
    mean_kernel<<<(B_ * D_ + 255) / 256, 256, 0, stream>>>(outSecond, feature);
    cls_kernel<<<1, 128, 0, stream>>>(feature, Wc, bc, cls);
}
